// LatentCNF_50242527428861
// MI455X (gfx1250) — 8-bit matrix operand order repaired and checked by reading, compiled, not run on hardware
//
#include <hip/hip_runtime.h>
#include <stdint.h>

#define LAT 64
#define HID 256
#define NSTEPS 32
#define BATCH 16384
#define ET 16
#define SSTR 72

typedef int   v8i  __attribute__((ext_vector_type(8)));
typedef int   v16i __attribute__((ext_vector_type(16)));
typedef float v8f  __attribute__((ext_vector_type(8)));

__device__ inline float tanh_fast(float x) {
  float xc = fminf(fmaxf(x, -9.0f), 9.0f);
  float t  = __builtin_amdgcn_exp2f(xc * 2.885390081777927f);
  float r  = __builtin_amdgcn_rcpf(t + 1.0f);
  return fmaf(-2.0f, r, 1.0f);
}

__device__ inline unsigned f32_to_e4m3_sw(float x) {
  unsigned sign = (__float_as_uint(x) >> 24) & 0x80u;
  float a = fminf(fabsf(x), 448.0f);
  unsigned u = __float_as_uint(a);
  unsigned r = u + 0x7FFFFu + ((u >> 20) & 1u);
  int e = (int)(r >> 23) - 127;
  unsigned norm = ((unsigned)(e + 7) << 3) | ((r >> 20) & 7u);
  norm = (norm > 0x7Eu) ? 0x7Eu : norm;
  unsigned den = (unsigned)(int)fmaf(a, 512.0f, 0.5f);
  unsigned code = (e < -6) ? den : norm;
  return sign | code;
}

__device__ inline unsigned pk_fp8(float a, float b) {
  return (unsigned)__builtin_amdgcn_cvt_pk_fp8_f32(a, b, 0, false);
}

__device__ inline float bf16_to_f32(unsigned h) { return __uint_as_float(h << 16); }
__device__ inline unsigned f32_to_bf16(float x) {
  unsigned u = __float_as_uint(x);
  return (u + 0x7FFFu + ((u >> 16) & 1u)) >> 16;
}

__device__ inline v16i loadA128(const char* bp, int rs, int m0, int k0, int lane) {
  int M = m0 + (lane & 15);
  int o = (lane & 16) ? 8 : 0;
  const char* p = bp + M * rs + k0 + o;
  union { v16i v; int2 c2[8]; } u;
#pragma unroll
  for (int c = 0; c < 8; ++c) u.c2[c] = *(const int2*)(p + 16 * c);
  return u.v;
}
__device__ inline v8i loadA64(const char* bp, int rs, int m0, int k0, int lane) {
  int M = m0 + (lane & 15);
  int o = (lane & 16) ? 8 : 0;
  const char* p = bp + M * rs + k0 + o;
  union { v8i v; int2 c2[4]; } u;
#pragma unroll
  for (int c = 0; c < 4; ++c) u.c2[c] = *(const int2*)(p + 16 * c);
  return u.v;
}
__device__ inline v16i loadB128(const char* bp, int rs, int k0, int n0, int lane) {
  union { v16i v; char b[64]; } u;
  int o = (lane & 16) ? 8 : 0;
#pragma unroll
  for (int s = 0; s < 64; ++s)
    u.b[s] = bp[(k0 + 16 * (s >> 3) + o + (s & 7)) * rs + n0 + (lane & 15)];
  return u.v;
}
__device__ inline v8i loadB64(const char* bp, int rs, int k0, int n0, int lane) {
  union { v8i v; char b[32]; } u;
  int o = (lane & 16) ? 8 : 0;
#pragma unroll
  for (int s = 0; s < 32; ++s)
    u.b[s] = bp[(k0 + 16 * (s >> 3) + o + (s & 7)) * rs + n0 + (lane & 15)];
  return u.v;
}

__global__ void cnf_prep(const float* __restrict__ W1, const float* __restrict__ W2,
                         const float* __restrict__ W3, const float* __restrict__ W4,
                         unsigned char* __restrict__ wq) {
  int i = blockIdx.x * blockDim.x + threadIdx.x;
  if (i >= 196608) return;
  if (i < 163840) {
    float v;
    if (i < 16384)       { int m = i >> 8, j = i & 255;                   v = W1[j * 64 + m]; }
    else if (i < 81920)  { int t = i - 16384;  int k = t >> 8, n = t & 255; v = W2[n * 256 + k]; }
    else if (i < 147456) { int t = i - 81920;  int k = t >> 8, n = t & 255; v = W3[n * 256 + k]; }
    else                 { int t = i - 147456; int k = t >> 6, n = t & 63;  v = W4[n * 256 + k]; }
    wq[i] = (unsigned char)f32_to_e4m3_sw(64.0f * v);
  } else {
    int t = i - 163840;
    unsigned short* wh = (unsigned short*)(wq + 163840);
    float v;
    if (t < 16384) { int m = t >> 8, j = t & 255; v = W1[j * 64 + m]; }
    else           { int q = t - 16384; int j = q >> 6, m = q & 63; v = W4[m * 256 + j]; }
    wh[t] = (unsigned short)f32_to_bf16(v);
  }
}

__global__ __launch_bounds__(256)
void cnf_main(const float* __restrict__ zin,
              const float* __restrict__ b1g, const float* __restrict__ b2g,
              const float* __restrict__ b3g, const float* __restrict__ b4g,
              const unsigned char* __restrict__ wq,
              float* __restrict__ out) {
  extern __shared__ char sm[];
  const int tid  = threadIdx.x;
  const int lane = tid & 31;
  const int wv   = tid >> 5;

  char*  sW1T8 = sm + 0;
  char*  sW2T8 = sm + 16384;
  char*  sW3T8 = sm + 81920;
  char*  sW4T8 = sm + 147456;
  char*  sW1Tb = sm + 163840;
  char*  sW4Tb = sm + 196608;
  char*  sV8   = sm + 229376;
  char*  sZ8   = sm + 245760;
  char*  sH8A  = sm + 246784;
  char*  sH8B  = sm + 250880;
  unsigned short* sD16 = (unsigned short*)(sm + 254976);
  float* sS   = (float*)(sm + 279552);
  float* sK   = (float*)(sm + 284160);
  float* sAcc = (float*)(sm + 288768);
  float* sB1  = (float*)(sm + 293376);
  float* sB2  = (float*)(sm + 294400);
  float* sB3  = (float*)(sm + 295424);
  float* sB4  = (float*)(sm + 296448);
  float* sDiv = (float*)(sm + 296704);

  {
    for (int i = tid; i < 229376 / 16; i += 256) {
      unsigned off = (unsigned)(i * 16);
      asm volatile("global_load_async_to_lds_b128 %0, %1, %2"
                   :
                   : "v"(off), "v"(off), "s"(wq)
                   : "memory");
    }
    asm volatile("s_wait_asynccnt 0" ::: "memory");
  }
  sB1[tid] = b1g[tid];
  sB2[tid] = b2g[tid];
  sB3[tid] = b3g[tid];
  if (tid < 64) sB4[tid] = b4g[tid];

  const int ebase = blockIdx.x * ET;
  for (int i = tid; i < ET * LAT; i += 256) {
    int e = i >> 6, m = i & 63;
    sS[e * SSTR + m] = zin[(ebase + e) * LAT + m];
  }
  if (tid < ET) sS[tid * SSTR + 64] = 0.0f;
  __syncthreads();

  const float dt = -1.0f / 32.0f;

#pragma unroll 1
  for (int step = 0; step < NSTEPS; ++step) {
#pragma unroll 1
    for (int st = 0; st < 4; ++st) {
      const float cc = (st == 0) ? 0.0f : ((st == 3) ? 1.0f : 0.5f);

      {
        int idx = tid * 4;
        int e = idx >> 6, m = idx & 63;
        float z4[4];
#pragma unroll
        for (int q = 0; q < 4; ++q) {
          float kz = (st == 0) ? 0.0f : sK[e * SSTR + m + q];
          z4[q] = sS[e * SSTR + m + q] + cc * dt * kz;
          sK[e * SSTR + m + q] = 0.0f;
        }
        unsigned lo = pk_fp8(z4[0], z4[1]) & 0xFFFFu;
        unsigned hi = pk_fp8(z4[2], z4[3]) & 0xFFFFu;
        *(unsigned*)(sZ8 + idx) = lo | (hi << 16);
      }
      if (tid < ET) sDiv[tid] = 0.0f;
      __syncthreads();

      {
        v8i A = loadA64(sZ8, LAT, 0, 0, lane);
#pragma unroll
        for (int t = 0; t < 2; ++t) {
          int n0 = (wv * 2 + t) * 16;
          v8i B = loadB64(sW1T8, HID, 0, n0, lane);
          v8f acc = {};
          acc = __builtin_amdgcn_wmma_f32_16x16x64_fp8_fp8(A, B, (short)0, acc, false, false);
          int n = n0 + (lane & 15);
#pragma unroll
          for (int r = 0; r < 8; r += 2) {
            int e0 = r + 8 * (lane >> 4);
            float h0 = tanh_fast(acc[r] * (1.0f / 64.0f) + sB1[n]);
            float h1 = tanh_fast(acc[r + 1] * (1.0f / 64.0f) + sB1[n]);
            sD16[e0 * HID + n]       = (unsigned short)f32_to_bf16(1.0f - h0 * h0);
            sD16[(e0 + 1) * HID + n] = (unsigned short)f32_to_bf16(1.0f - h1 * h1);
            unsigned pk = pk_fp8(h0, h1);
            ((unsigned char*)sH8A)[e0 * HID + n]       = (unsigned char)(pk & 0xFFu);
            ((unsigned char*)sH8A)[(e0 + 1) * HID + n] = (unsigned char)((pk >> 8) & 0xFFu);
          }
        }
      }
      __syncthreads();

#pragma unroll 1
      for (int ly = 0; ly < 2; ++ly) {
        const char* Wb   = ly ? sW3T8 : sW2T8;
        const char* Ain  = ly ? sH8B : sH8A;
        char*       Aout = ly ? sH8A : sH8B;
        const float* bias = ly ? sB3 : sB2;
        unsigned short* drow = sD16 + (ly + 1) * ET * HID;
        v16i A0 = loadA128(Ain, HID, 0, 0, lane);
        v16i A1 = loadA128(Ain, HID, 0, 128, lane);
#pragma unroll
        for (int t = 0; t < 2; ++t) {
          int n0 = (wv * 2 + t) * 16;
          v8f acc = {};
          v16i B0 = loadB128(Wb, HID, 0, n0, lane);
          acc = __builtin_amdgcn_wmma_f32_16x16x128_fp8_fp8(A0, B0, (short)0, acc, false, false);
          v16i B1 = loadB128(Wb, HID, 128, n0, lane);
          acc = __builtin_amdgcn_wmma_f32_16x16x128_fp8_fp8(A1, B1, (short)0, acc, false, false);
          int n = n0 + (lane & 15);
#pragma unroll
          for (int r = 0; r < 8; r += 2) {
            int e0 = r + 8 * (lane >> 4);
            float h0 = tanh_fast(acc[r] * (1.0f / 64.0f) + bias[n]);
            float h1 = tanh_fast(acc[r + 1] * (1.0f / 64.0f) + bias[n]);
            drow[e0 * HID + n]       = (unsigned short)f32_to_bf16(1.0f - h0 * h0);
            drow[(e0 + 1) * HID + n] = (unsigned short)f32_to_bf16(1.0f - h1 * h1);
            unsigned pk = pk_fp8(h0, h1);
            ((unsigned char*)Aout)[e0 * HID + n]       = (unsigned char)(pk & 0xFFu);
            ((unsigned char*)Aout)[(e0 + 1) * HID + n] = (unsigned char)((pk >> 8) & 0xFFu);
          }
        }
        __syncthreads();
      }

      {
        int nt = wv & 3;
        int kb = wv >> 2;
        v16i A = loadA128(sH8A, HID, 0, kb * 128, lane);
        v16i B = loadB128(sW4T8, LAT, kb * 128, nt * 16, lane);
        v8f acc = {};
        acc = __builtin_amdgcn_wmma_f32_16x16x128_fp8_fp8(A, B, (short)0, acc, false, false);
#pragma unroll
        for (int r = 0; r < 8; ++r) {
          int e = r + 8 * (lane >> 4);
          int n = nt * 16 + (lane & 15);
          float v = acc[r] * (1.0f / 64.0f) + ((kb == 0) ? sB4[n] : 0.0f);
          atomicAdd(&sK[e * SSTR + n], v);
        }
      }
      __syncthreads();

#pragma unroll 1
      for (int e = 0; e < ET; ++e) {
        const unsigned short* d1 = sD16 + e * HID;
        const unsigned short* d2 = sD16 + (ET + e) * HID;
        const unsigned short* d3 = sD16 + (2 * ET + e) * HID;

        {
          int j0 = (tid & 127) * 2;
          int m0 = (tid >> 7) * 32;
          unsigned dpair = *(const unsigned*)(d1 + j0);
          float s0 = bf16_to_f32(dpair & 0xFFFFu) * 4096.0f;
          float s1 = bf16_to_f32(dpair >> 16) * 4096.0f;
#pragma unroll 1
          for (int m = m0; m < m0 + 32; ++m) {
            unsigned w2 = *(const unsigned*)(sW1Tb + m * 512 + j0 * 2);
            float a = bf16_to_f32(w2 & 0xFFFFu) * s0;
            float b = bf16_to_f32(w2 >> 16) * s1;
            *(unsigned short*)(sV8 + m * HID + j0) = (unsigned short)pk_fp8(a, b);
          }
        }
        __syncthreads();

        {
          int mt  = wv >> 1;
          int nt0 = (wv & 1) * 8;
          v8f acc[8] = {};
#pragma unroll
          for (int kb = 0; kb < 2; ++kb) {
            v16i A = loadA128(sV8, HID, mt * 16, kb * 128, lane);
#pragma unroll
            for (int t = 0; t < 8; ++t) {
              v16i B = loadB128(sW2T8, HID, kb * 128, (nt0 + t) * 16, lane);
              acc[t] = __builtin_amdgcn_wmma_f32_16x16x128_fp8_fp8(A, B, (short)0,
                                                                   acc[t], false, false);
            }
          }
          __syncthreads();
#pragma unroll
          for (int t = 0; t < 8; ++t) {
            int n = (nt0 + t) * 16 + (lane & 15);
            float dd = bf16_to_f32(d2[n]) * (1.0f / 64.0f);
#pragma unroll
            for (int r = 0; r < 8; r += 2) {
              int m = mt * 16 + r + 8 * (lane >> 4);
              unsigned pk = pk_fp8(acc[t][r] * dd, acc[t][r + 1] * dd);
              ((unsigned char*)sV8)[m * HID + n]       = (unsigned char)(pk & 0xFFu);
              ((unsigned char*)sV8)[(m + 1) * HID + n] = (unsigned char)((pk >> 8) & 0xFFu);
            }
          }
        }
        __syncthreads();

        {
          int mt  = wv >> 1;
          int nt0 = (wv & 1) * 8;
          v8f acc[8] = {};
#pragma unroll
          for (int kb = 0; kb < 2; ++kb) {
            v16i A = loadA128(sV8, HID, mt * 16, kb * 128, lane);
#pragma unroll
            for (int t = 0; t < 8; ++t) {
              v16i B = loadB128(sW3T8, HID, kb * 128, (nt0 + t) * 16, lane);
              acc[t] = __builtin_amdgcn_wmma_f32_16x16x128_fp8_fp8(A, B, (short)0,
                                                                   acc[t], false, false);
            }
          }
          float p = 0.0f;
          int m0 = mt * 16 + 8 * (lane >> 4);
#pragma unroll
          for (int t = 0; t < 8; ++t) {
            int n = (nt0 + t) * 16 + (lane & 15);
            float dd = bf16_to_f32(d3[n]);
            int4 wt = *(const int4*)(sW4Tb + n * 128 + m0 * 2);
            unsigned wa[4] = {(unsigned)wt.x, (unsigned)wt.y,
                              (unsigned)wt.z, (unsigned)wt.w};
#pragma unroll
            for (int r = 0; r < 8; ++r) {
              unsigned hw = (wa[r >> 1] >> ((r & 1) * 16)) & 0xFFFFu;
              p += acc[t][r] * dd * bf16_to_f32(hw);
            }
          }
          p += __shfl_xor(p, 1);
          p += __shfl_xor(p, 2);
          p += __shfl_xor(p, 4);
          p += __shfl_xor(p, 8);
          p += __shfl_xor(p, 16);
          if (lane == 0) atomicAdd(&sDiv[e], p);
        }
        __syncthreads();
      }

      if (tid < ET) sK[tid * SSTR + 64] = -sDiv[tid] * (1.0f / 262144.0f);
      __syncthreads();

      {
        const float w = (st == 1 || st == 2) ? 2.0f : 1.0f;
        for (int i = tid; i < ET * 65; i += 256) {
          int e = i / 65, q = i % 65;
          float k = sK[e * SSTR + q];
          if (st == 0)      sAcc[e * SSTR + q] = k;
          else if (st < 3)  sAcc[e * SSTR + q] += w * k;
          else              sS[e * SSTR + q] += (dt / 6.0f) * (sAcc[e * SSTR + q] + k);
        }
      }
      __syncthreads();
    }
  }

  for (int i = tid; i < ET * LAT; i += 256) {
    int e = i >> 6, m = i & 63;
    out[(ebase + e) * LAT + m] = sS[e * SSTR + m];
  }
  if (tid < ET) out[BATCH * LAT + ebase + tid] = sS[tid * SSTR + 64];
}

extern "C" void kernel_launch(void* const* d_in, const int* in_sizes, int n_in,
                              void* d_out, int out_size, void* d_ws, size_t ws_size,
                              hipStream_t stream) {
  (void)in_sizes; (void)n_in; (void)out_size; (void)ws_size;
  const float* z  = (const float*)d_in[0];
  const float* W1 = (const float*)d_in[1];
  const float* b1 = (const float*)d_in[2];
  const float* W2 = (const float*)d_in[3];
  const float* b2 = (const float*)d_in[4];
  const float* W3 = (const float*)d_in[5];
  const float* b3 = (const float*)d_in[6];
  const float* W4 = (const float*)d_in[7];
  const float* b4 = (const float*)d_in[8];
  unsigned char* wq = (unsigned char*)d_ws;

  cnf_prep<<<(196608 + 255) / 256, 256, 0, stream>>>(W1, W2, W3, W4, wq);

  const size_t ldsBytes = 296768;
  cnf_main<<<BATCH / ET, 256, ldsBytes, stream>>>(z, b1, b2, b3, b4, wq,
                                                  (float*)d_out);
}
